// PredictionLayer_56238301774541
// MI455X (gfx1250) — matrix operand order repaired and checked by reading, not compiled, not run on hardware
//
#include <hip/hip_runtime.h>
#include <hip/hip_bf16.h>

typedef __attribute__((ext_vector_type(16))) _Float16 v16h;
typedef __attribute__((ext_vector_type(8)))  float    v8f;

#define NB      2
#define NNODES  1024
#define FDIM    64
#define SLOPE   0.01f

// Position of logical K index k inside the pre-swizzled 64-half row so that
// lane-group g of a wave reads its 16-bit A-matrix fragment halves
// contiguously: [g*32 .. g*32+15] = frag0 (K<32), [g*32+16 .. g*32+31] = frag1.
__device__ __forceinline__ int frag_pos(int k) {
  return ((k >> 3) & 1) * 32   // lane group (K bit 3)
       + ((k >> 5) & 1) * 16   // K-half fragment (K bit 5)
       + ((k >> 4) & 1) * 8    // idx 0-7 vs 8-15 inside fragment (K bit 4)
       + (k & 7);
}

// ---------------------------------------------------------------------------
// Kernel 1: left = emb @ W1[:F] + b1 ; right = emb @ W1[F:]
// Stored as f16 in WMMA-A swizzled order. One 64-thread block per row. Tiny.
// ---------------------------------------------------------------------------
__global__ __launch_bounds__(FDIM) void precompute_kernel(
    const float* __restrict__ emb, const float* __restrict__ W1,
    const float* __restrict__ b1, _Float16* __restrict__ leftb,
    _Float16* __restrict__ rightb) {
  __shared__ float se[FDIM];
  const int row = blockIdx.x;          // 0 .. 2*N-1
  const int c   = threadIdx.x;         // 0 .. 63  (output feature = K index)
  se[c] = emb[(size_t)row * FDIM + c];
  __syncthreads();
  float l = b1[c];                     // fold b1 into left once
  float r = 0.f;
#pragma unroll 8
  for (int k = 0; k < FDIM; ++k) {
    const float e = se[k];
    l += e * W1[k * FDIM + c];
    r += e * W1[(FDIM + k) * FDIM + c];
  }
  const int pos = frag_pos(c);
  leftb [(size_t)row * FDIM + pos] = (_Float16)l;
  rightb[(size_t)row * FDIM + pos] = (_Float16)r;
}

__device__ __forceinline__ float leakyf(float x) {
  return fmaxf(x, SLOPE * x);   // slope < 1 -> max form
}

// ---------------------------------------------------------------------------
// Kernel 2: fused pairwise MLP. One wave owns a 16-wide j tile (WMMA M dim)
// and iterates over 64 i rows. 4 x v_wmma_f32_16x16x32_f16 per output row,
// packed-f16 activation, 8-shuffle reduce-scatter epilogue.
// ---------------------------------------------------------------------------
__global__ __launch_bounds__(256) void edge_mlp_kernel(
    const _Float16* __restrict__ leftb, const _Float16* __restrict__ rightb,
    const float* __restrict__ W2, const float* __restrict__ b2v,
    const float* __restrict__ W3, const float* __restrict__ b3v,
    float* __restrict__ out) {
  const int lane = threadIdx.x & 31;
  const int n    = lane & 15;          // output channel / B-matrix N index
  const int g    = lane >> 4;          // lane group
  const int wid  = blockIdx.x * (blockDim.x >> 5) + (threadIdx.x >> 5);

  // wave decomposition: 2 batches * 64 j-tiles * 16 i-chunks = 2048 waves
  const int waves_per_b = (NNODES / 16) * (NNODES / 64);
  const int b   = wid / waves_per_b;
  const int rem = wid % waves_per_b;
  const int jt  = rem / (NNODES / 64);
  const int ic  = rem % (NNODES / 64);
  const int j0      = jt * 16;
  const int i_begin = ic * 64;

  // --- W2 (64x32 f32) -> four f16 B fragments, resident for the whole wave.
  // B layout (16-bit, 32x16): lane group g holds K = 8g+idx (idx<8), 16+8g+(idx-8) (idx>=8), N = lane&15.
  v16h bfrag[2][2];
#pragma unroll
  for (int H = 0; H < 2; ++H)
#pragma unroll
    for (int nh = 0; nh < 2; ++nh) {
      v16h t;
#pragma unroll
      for (int k = 0; k < 16; ++k)
        t[k] = (_Float16)W2[(32 * H + 8 * g + (k & 7) + 16 * (k >> 3)) * 32 + 16 * nh + n];
      bfrag[H][nh] = t;
    }

  const float b2n0 = b2v[n], b2n1 = b2v[16 + n];
  const float w3n0 = W3[n],  w3n1 = W3[16 + n];
  const float b3s  = b3v[0];

  // --- right tile (A rows = 16 j values), pre-swizzled f16: direct v16h loads
  const v16h* rrow = reinterpret_cast<const v16h*>(
      rightb + ((size_t)b * NNODES + j0 + n) * FDIM + 32 * g);
  const v16h rg0 = rrow[0];
  const v16h rg1 = rrow[1];

  const _Float16* lbase = leftb + (size_t)b * NNODES * FDIM + 32 * g;
  const _Float16 hslope = (_Float16)SLOPE;

  for (int i = i_begin; i < i_begin + 64; ++i) {
    const v16h* lrow = reinterpret_cast<const v16h*>(lbase + (size_t)i * FDIM);
    __builtin_prefetch((const void*)(lbase + (size_t)(i + 1) * FDIM), 0, 1);

    // A fragments: h = leaky(left + right), all packed f16 (v_pk_* ops)
    const v16h x0 = lrow[0] + rg0;
    const v16h x1 = lrow[1] + rg1;
    const v16h a0 = __builtin_elementwise_max(x0, x0 * hslope);
    const v16h a1 = __builtin_elementwise_max(x1, x1 * hslope);

    v8f c0 = {};
    v8f c1 = {};
    c0 = __builtin_amdgcn_wmma_f32_16x16x32_f16(false, a0, false, bfrag[0][0],
                                                (short)0, c0, false, false);
    c0 = __builtin_amdgcn_wmma_f32_16x16x32_f16(false, a1, false, bfrag[1][0],
                                                (short)0, c0, false, false);
    c1 = __builtin_amdgcn_wmma_f32_16x16x32_f16(false, a0, false, bfrag[0][1],
                                                (short)0, c1, false, false);
    c1 = __builtin_amdgcn_wmma_f32_16x16x32_f16(false, a1, false, bfrag[1][1],
                                                (short)0, c1, false, false);

    // Per-channel epilogue: s[r] = leaky(D+b2) . W3 partial for row r.
    float s[8];
#pragma unroll
    for (int r = 0; r < 8; ++r)
      s[r] = leakyf(c0[r] + b2n0) * w3n0 + leakyf(c1[r] + b2n1) * w3n1;

    // Reduce-scatter butterfly over the 16-lane group: 8 shuffles total.
    // After masks 8,4,2 each lane holds one r reduced over 8 lanes; final
    // mask-1 step completes the 16-lane sum. Lane n ends with r=(n>>1)&7.
#pragma unroll
    for (int q = 0; q < 4; ++q) {
      const float send = (n & 8) ? s[q] : s[4 + q];
      const float recv = __shfl_xor(send, 8, 32);
      s[q] = ((n & 8) ? s[4 + q] : s[q]) + recv;
    }
#pragma unroll
    for (int q = 0; q < 2; ++q) {
      const float send = (n & 4) ? s[q] : s[2 + q];
      const float recv = __shfl_xor(send, 4, 32);
      s[q] = ((n & 4) ? s[2 + q] : s[q]) + recv;
    }
    {
      const float send = (n & 2) ? s[0] : s[1];
      const float recv = __shfl_xor(send, 2, 32);
      s[0] = ((n & 2) ? s[1] : s[0]) + recv;
    }
    s[0] += __shfl_xor(s[0], 1, 32);

    // Even lanes of each group store 8 contiguous outputs (j = j0+8g+r).
    const int r = (n >> 1) & 7;
    if ((n & 1) == 0)
      out[((size_t)b * NNODES + i) * NNODES + j0 + 8 * g + r] = s[0] + b3s;
  }
}

extern "C" void kernel_launch(void* const* d_in, const int* in_sizes, int n_in,
                              void* d_out, int out_size, void* d_ws, size_t ws_size,
                              hipStream_t stream) {
  const float* emb = (const float*)d_in[0];
  const float* W1  = (const float*)d_in[1];
  const float* b1  = (const float*)d_in[2];
  const float* W2  = (const float*)d_in[3];
  const float* b2  = (const float*)d_in[4];
  const float* W3  = (const float*)d_in[5];
  const float* b3  = (const float*)d_in[6];
  float* out = (float*)d_out;

  _Float16* leftb  = (_Float16*)d_ws;                       // [2*1024*64] f16
  _Float16* rightb = leftb + (size_t)NB * NNODES * FDIM;    // [2*1024*64] f16

  precompute_kernel<<<NB * NNODES, FDIM, 0, stream>>>(emb, W1, b1, leftb, rightb);

  const int total_waves = NB * (NNODES / 16) * (NNODES / 64);  // 2048
  const int waves_per_block = 8;                               // 256 threads
  edge_mlp_kernel<<<total_waves / waves_per_block, waves_per_block * 32, 0,
                    stream>>>(leftb, rightb, W2, b2, W3, b3, out);
}
